// HybridEstimatorQLSTM_65481071398760
// MI455X (gfx1250) — matrix operand order repaired and checked by reading, not compiled, not run on hardware
//
#include <hip/hip_runtime.h>
#include <hip/hip_bf16.h>

typedef __attribute__((ext_vector_type(16))) _Float16 v16h;
typedef __attribute__((ext_vector_type(8)))  float    v8f;

#define BQ   4096          // batch
#define SQ   128           // seq len
#define DQ   64            // input dim
#define HQ   8             // hidden / n_qubits
#define ROWS (BQ * SQ)     // GEMM M
#define NW   32            // GEMM N (4 gates * 8 units)
#define XW_BYTES ((size_t)ROWS * NW * sizeof(float))   // 64 MB z buffer

// ---------------------------------------------------------------------------
// Setup: pack the x-part of the 4 gate weights (rows 0..63 of each (72,8))
// into the exact f16 B-matrix lane layout for v_wmma_f32_16x16x32_f16:
//   B 32x16: lanes 0-15 hold K=0..7,16..23 (2 halves/VGPR), lanes 16-31 K=8..15,24..31,
//   column N = lane%16.  Index (c*2+t): c = K-chunk (0:K0-31,1:K32-63),
//   t = N-tile (0:j0-15, 1:j16-31).  Also fold bias+theta into biasc[32].
// ---------------------------------------------------------------------------
__global__ void pack_weights(const float* __restrict__ Wf, const float* __restrict__ Wi,
                             const float* __restrict__ Wu, const float* __restrict__ Wo,
                             const float* __restrict__ bf, const float* __restrict__ bi,
                             const float* __restrict__ bu, const float* __restrict__ bo,
                             const float* __restrict__ tf, const float* __restrict__ ti,
                             const float* __restrict__ tu, const float* __restrict__ to_,
                             _Float16* __restrict__ Wpack, float* __restrict__ biasc) {
  int tid  = threadIdx.x;          // 128 threads: (c,t,lane)
  int lane = tid & 31;
  int ct   = tid >> 5;             // c*2 + t
  int t    = ct & 1;
  int c    = ct >> 1;
  int N    = 16 * t + (lane & 15); // combined column j = g*8 + u
  int g    = N >> 3, u = N & 7;
  const float* Wg = (g == 0) ? Wf : (g == 1) ? Wi : (g == 2) ? Wu : Wo;
  int kbase = 32 * c + 8 * (lane >> 4);
  #pragma unroll
  for (int h = 0; h < 16; ++h) {
    int K = kbase + h + (h >= 8 ? 8 : 0);    // 0..63 => x-part rows of (72,8)
    Wpack[(ct * 32 + lane) * 16 + h] = (_Float16)Wg[K * 8 + u];
  }
  if (tid < 32) {
    int g2 = tid >> 3, u2 = tid & 7;
    const float* bg = (g2 == 0) ? bf : (g2 == 1) ? bi : (g2 == 2) ? bu : bo;
    const float* tg = (g2 == 0) ? tf : (g2 == 1) ? ti : (g2 == 2) ? tu : to_;
    biasc[tid] = bg[u2] + tg[u2];
  }
}

// ---------------------------------------------------------------------------
// Phase 1: z[m][j] = x_row(m) . Wx[:,j] + bias[j] + theta[j]
// M = 524288, K = 64, N = 32.  One wave per 16x32 output tile:
// 2 K-chunk WMMAs per N-tile = 4 v_wmma_f32_16x16x32_f16 per wave.
// Epilogue stores z with columns reordered to [unit][gate] so the scan
// kernel's per-lane float4 load grabs all 4 gates of its hidden unit.
// ---------------------------------------------------------------------------
__global__ __launch_bounds__(256) void gemm_xw(const float* __restrict__ x,
                                               const _Float16* __restrict__ Wpack,
                                               const float* __restrict__ biasc,
                                               float* __restrict__ xW) {
  int tid  = threadIdx.x;
  int lane = tid & 31;
  int wv   = tid >> 5;                                // 8 waves / block
  long rowbase = ((long)blockIdx.x * 8 + wv) * 16;    // 16 rows of x per wave
  int  hi   = lane >> 4;
  long row  = rowbase + (lane & 15);
  const float* xr = x + row * DQ;
  int  koff = hi ? 8 : 0;

  // A tiles (16x32 f16): halves 0..7 = K koff..koff+7, halves 8..15 = +16
  v16h a0, a1;
  #pragma unroll
  for (int q = 0; q < 8; ++q) {
    a0[q]     = (_Float16)xr[koff + q];
    a0[q + 8] = (_Float16)xr[koff + 16 + q];
    a1[q]     = (_Float16)xr[32 + koff + q];
    a1[q + 8] = (_Float16)xr[32 + koff + 16 + q];
  }

  // init accumulators with bias+theta (C element depends only on column N)
  int n = lane & 15;
  float b0v = biasc[n], b1v = biasc[n + 16];
  v8f acc0, acc1;
  #pragma unroll
  for (int v = 0; v < 8; ++v) { acc0[v] = b0v; acc1[v] = b1v; }

  const v16h* Wp = (const v16h*)Wpack;
  v8f d0 = __builtin_amdgcn_wmma_f32_16x16x32_f16(false, a0, false, Wp[0 * 32 + lane],
                                                  (short)0, acc0, false, false);
  d0     = __builtin_amdgcn_wmma_f32_16x16x32_f16(false, a1, false, Wp[2 * 32 + lane],
                                                  (short)0, d0,   false, false);
  v8f d1 = __builtin_amdgcn_wmma_f32_16x16x32_f16(false, a0, false, Wp[1 * 32 + lane],
                                                  (short)0, acc1, false, false);
  d1     = __builtin_amdgcn_wmma_f32_16x16x32_f16(false, a1, false, Wp[3 * 32 + lane],
                                                  (short)0, d1,   false, false);

  // C layout: VGPR v, lane L -> M = v + 8*(L/16), N = L%16.
  // Reorder column j = g*8+u  ->  u*4+g for the scan kernel.
  int col0 = (n & 7) * 4 + (n >> 3);          // j = n      (gates 0,1)
  int col1 = (n & 7) * 4 + 2 + (n >> 3);      // j = n + 16 (gates 2,3)
  #pragma unroll
  for (int v = 0; v < 8; ++v) {
    long m = rowbase + v + 8 * hi;
    xW[m * NW + col0] = d0[v];
    xW[m * NW + col1] = d1[v];
  }
}

// ---------------------------------------------------------------------------
// Octet (8-lane) helpers for the recurrent scan
// ---------------------------------------------------------------------------
__device__ __forceinline__ float lanebc(float v, int srcLane) {
  return __int_as_float(__builtin_amdgcn_ds_bpermute(srcLane << 2, __float_as_int(v)));
}
__device__ __forceinline__ float fsig(float x)  { return 1.f / (1.f + __expf(-x)); }
__device__ __forceinline__ float ftanh(float x) {
  float cx = fminf(fmaxf(x, -15.f), 15.f);
  float e  = __expf(-2.f * cx);
  return (1.f - e) / (1.f + e);
}
// qgate over one octet: lane u holds phi_u; returns <Z_u>:
//   u>=1: prod_{0..u} cos(phi_i);  u==0: prod_{1..7} cos(phi_i)
__device__ __forceinline__ float octet_qgate(float phi, int lane, int u) {
  float c = __cosf(phi);
  float q = (u == 0) ? 1.f : c;          // prefix over lanes 1..7 only
  float p = q;
  float t1 = lanebc(p, lane - 1); p = (u >= 1) ? p * t1 : p;
  float t2 = lanebc(p, lane - 2); p = (u >= 2) ? p * t2 : p;
  float t4 = lanebc(p, lane - 4); p = (u >= 4) ? p * t4 : p;
  int ob   = lane & ~7;
  float c0  = lanebc(c, ob);             // cos(phi_0)
  float tot = lanebc(p, ob + 7);         // prod_{1..7}
  return (u == 0) ? tot : c0 * p;
}

// ---------------------------------------------------------------------------
// Phase 2: LSTM scan.  8 lanes per batch element (one per hidden unit) =
// 32768 threads / 1024 waves -> fills the machine far better than 1 thr/b.
// Recurrent weights (32 floats per lane) live in registers; h exchanged via
// ds_bpermute; cumprod is a 3-step octet prefix.  No LDS in the hot loop.
// ---------------------------------------------------------------------------
__global__ __launch_bounds__(256) void qlstm_scan(const float* __restrict__ xW,
                                                  const float* __restrict__ Wf,
                                                  const float* __restrict__ Wi,
                                                  const float* __restrict__ Wu,
                                                  const float* __restrict__ Wo,
                                                  const float* __restrict__ Wt,
                                                  const float* __restrict__ bt,
                                                  float* __restrict__ tag) {
  int  tid  = threadIdx.x;
  int  lane = tid & 31;
  long gid  = (long)blockIdx.x * 256 + tid;
  long b    = gid >> 3;                  // batch element
  int  u    = lane & 7;                  // hidden unit handled by this lane

  // recurrent weights: wh[uh][g] = W_g[(64+uh)*8 + u]
  float wh[8][4];
  #pragma unroll
  for (int uh = 0; uh < 8; ++uh) {
    int r = (64 + uh) * 8 + u;
    wh[uh][0] = Wf[r]; wh[uh][1] = Wi[r]; wh[uh][2] = Wu[r]; wh[uh][3] = Wo[r];
  }
  float wtu = Wt[u];
  float bt0 = bt[0];

  float h = 0.f, cst = 0.f;
  const float4* zbase = (const float4*)(xW + (size_t)b * SQ * NW);

  for (int s = 0; s < SQ; ++s) {
    float4 z = zbase[s * 8 + u];         // 4 gate pre-acts of unit u (from x)
    float zf = z.x, zi = z.y, zu = z.z, zo = z.w;
    #pragma unroll
    for (int uh = 0; uh < 8; ++uh) {     // + h_{t-1} @ W_h
      float hv = lanebc(h, (lane & ~7) + uh);
      zf += hv * wh[uh][0]; zi += hv * wh[uh][1];
      zu += hv * wh[uh][2]; zo += hv * wh[uh][3];
    }
    float f = fsig (octet_qgate(zf, lane, u));
    float i = fsig (octet_qgate(zi, lane, u));
    float g = ftanh(octet_qgate(zu, lane, u));
    float o = fsig (octet_qgate(zo, lane, u));
    cst = f * cst + i * g;
    h   = o * ftanh(cst);

    float tg = h * wtu;                  // octet butterfly reduction
    tg += lanebc(tg, lane ^ 1);
    tg += lanebc(tg, lane ^ 2);
    tg += lanebc(tg, lane ^ 4);
    if (u == 0) tag[b * SQ + s] = tg + bt0;
  }
}

// ---------------------------------------------------------------------------
// Phase 3: regressor head on x[:,0,:]:  tanh(x0 @ Wr1 + br1) @ Wr2 + br2
// ---------------------------------------------------------------------------
__global__ __launch_bounds__(256) void reg_head(const float* __restrict__ x,
                                                const float* __restrict__ Wr1,
                                                const float* __restrict__ br1,
                                                const float* __restrict__ Wr2,
                                                const float* __restrict__ br2,
                                                float* __restrict__ out) {
  __shared__ __align__(16) float sW[DQ * HQ];
  __shared__ float sb1[HQ], sW2[HQ], sb2;
  int tid = threadIdx.x;
  sW[tid]       = Wr1[tid];
  sW[tid + 256] = Wr1[tid + 256];
  if (tid < HQ) { sb1[tid] = br1[tid]; sW2[tid] = Wr2[tid]; }
  if (tid == 0) sb2 = br2[0];
  __syncthreads();

  long b = (long)blockIdx.x * 256 + tid;
  const float* xr = x + b * (size_t)SQ * DQ;   // x[b,0,:]
  float acc[HQ];
  #pragma unroll
  for (int q = 0; q < HQ; ++q) acc[q] = sb1[q];
  for (int k = 0; k < DQ; ++k) {
    float xv = xr[k];
    #pragma unroll
    for (int q = 0; q < HQ; ++q) acc[q] += xv * sW[k * HQ + q];
  }
  float r = sb2;
  #pragma unroll
  for (int q = 0; q < HQ; ++q) r += ftanh(acc[q]) * sW2[q];
  out[b] = r;
}

// ---------------------------------------------------------------------------
extern "C" void kernel_launch(void* const* d_in, const int* in_sizes, int n_in,
                              void* d_out, int out_size, void* d_ws, size_t ws_size,
                              hipStream_t stream) {
  const float* x   = (const float*)d_in[0];
  const float* Wf  = (const float*)d_in[1];  const float* bf  = (const float*)d_in[2];
  const float* Wi  = (const float*)d_in[3];  const float* bi  = (const float*)d_in[4];
  const float* Wu  = (const float*)d_in[5];  const float* bu  = (const float*)d_in[6];
  const float* Wo  = (const float*)d_in[7];  const float* bo  = (const float*)d_in[8];
  const float* tf  = (const float*)d_in[9];  const float* ti  = (const float*)d_in[10];
  const float* tu  = (const float*)d_in[11]; const float* to_ = (const float*)d_in[12];
  const float* Wr1 = (const float*)d_in[13]; const float* br1 = (const float*)d_in[14];
  const float* Wr2 = (const float*)d_in[15]; const float* br2 = (const float*)d_in[16];
  const float* Wt  = (const float*)d_in[17]; const float* bt  = (const float*)d_in[18];

  char*      wsB   = (char*)d_ws;
  float*     xW    = (float*)wsB;                        // 64 MB z buffer
  _Float16*  Wpack = (_Float16*)(wsB + XW_BYTES);        // 4 KB packed B tiles
  float*     biasc = (float*)(wsB + XW_BYTES + 4096);    // 32 floats

  float* reg = (float*)d_out;         // (B,1)
  float* tag = (float*)d_out + BQ;    // (B,S,1)

  pack_weights<<<1, 128, 0, stream>>>(Wf, Wi, Wu, Wo, bf, bi, bu, bo,
                                      tf, ti, tu, to_, Wpack, biasc);
  gemm_xw<<<ROWS / (16 * 8), 256, 0, stream>>>(x, Wpack, biasc, xW);
  qlstm_scan<<<(BQ * HQ) / 256, 256, 0, stream>>>(xW, Wf, Wi, Wu, Wo, Wt, bt, tag);
  reg_head<<<BQ / 256, 256, 0, stream>>>(x, Wr1, br1, Wr2, br2, reg);
}
